// StructFormer_17325898072154
// MI455X (gfx1250) — matrix operand order repaired and checked by reading, not compiled, not run on hardware
//
#include <hip/hip_runtime.h>
#include <hip/hip_bf16.h>
#include <math.h>

// ---------------------------------------------------------------------------
// Types for CDNA5 WMMA (wave32, v_wmma_f32_16x16x32_bf16)
// ---------------------------------------------------------------------------
typedef __bf16 bf16_t;
typedef __attribute__((ext_vector_type(16))) __bf16 v16bf;
typedef __attribute__((ext_vector_type(8)))  __bf16 v8bf;
typedef __attribute__((ext_vector_type(4)))  __bf16 v4bf;
typedef __attribute__((ext_vector_type(8)))  float  v8f;

#define BM 64
#define BN 64
#define BK 32
#define LDSS 40      // BK + 8 bf16 pad -> 80B rows, 16B aligned
#define GT 128       // 4 waves of 32

// ---------------------------------------------------------------------------
// Generic bf16 WMMA GEMM:  C = act(alpha * A @ op(W) + bias) + res (+C if accum)
//   wLayout==0 : W is [N][K] row-major (i.e. X @ W.T)
//   wLayout==1 : W is [K][N] row-major
//   rowShift/segLen : A-row remap for shifted conv taps (zero-pad outside seg)
//   batching: base += (z/zdiv)*s?1 + (z%zdiv)*s?2
// Requires K % 32 == 0 (true for every GEMM in this net).
// ---------------------------------------------------------------------------
struct GemmP {
  const float* A; const float* W; const float* bias; const float* res; float* C;
  int M, N, K;
  int lda, ldw, ldc;
  int wLayout;
  float alpha;
  int act;          // 0 none, 1 leaky-relu(0.01)
  int accum;
  int rowShift, segLen;
  long sA1; int sA2;
  long sW1; int sW2;
  long sC1; int sC2;
  int zdiv;
};

__global__ __launch_bounds__(GT) void gemm_wmma_bf16(GemmP pp) {
  __shared__ bf16_t As[2][BM][LDSS];
  __shared__ bf16_t Bs[2][BN][LDSS];

  const int tid  = threadIdx.x;
  const int lane = tid & 31;
  const int wave = tid >> 5;
  const int wRow = wave >> 1;       // 0..1
  const int wCol = wave & 1;        // 0..1
  const int half = lane >> 4;       // 0..1
  const int l16  = lane & 15;

  const int n0 = blockIdx.x * BN;
  const int m0 = blockIdx.y * BM;
  const int z  = blockIdx.z;
  const int zq = z / pp.zdiv;
  const int zr = z - zq * pp.zdiv;

  const float* Ab = pp.A + zq * pp.sA1 + (long)zr * pp.sA2;
  const float* Wb = pp.W + zq * pp.sW1 + (long)zr * pp.sW2;
  float*       Cb = pp.C + zq * pp.sC1 + (long)zr * pp.sC2;

  // ---- hoisted, K-invariant addressing (advanced by fixed strides) ----
  int rA[4], cA[4];
  const float* aP[4];
#pragma unroll
  for (int it = 0; it < 4; ++it) {
    int idx = tid + it * GT;            // 0..511
    int r = idx >> 3, kq = idx & 7;
    rA[it] = r; cA[it] = kq * 4;
    const float* p = nullptr;
    int mg = m0 + r;
    if (mg < pp.M) {
      if (pp.rowShift == 0) {
        p = Ab + (long)mg * pp.lda + cA[it];
      } else {
        int bb = mg / pp.segLen;
        int ll = mg - bb * pp.segLen + pp.rowShift;
        if (ll >= 0 && ll < pp.segLen)
          p = Ab + ((long)bb * pp.segLen + ll) * pp.lda + cA[it];
      }
    }
    aP[it] = p;
  }

  int rW[4], cW[4];
  const float* wP[4];
  long wAdv;
  const bool edgeN = (n0 + BN) > pp.N;
  if (pp.wLayout == 0) {
    wAdv = BK;
#pragma unroll
    for (int it = 0; it < 4; ++it) {
      int idx = tid + it * GT;
      int n = idx >> 3, kq = idx & 7;
      rW[it] = n; cW[it] = kq * 4;
      int ng = n0 + n;
      wP[it] = (ng < pp.N) ? (Wb + (long)ng * pp.ldw + kq * 4) : nullptr;
    }
  } else {
    wAdv = (long)BK * pp.ldw;
#pragma unroll
    for (int it = 0; it < 4; ++it) {
      int idx = tid + it * GT;            // 32 k-rows x 16 n-quads
      int k = idx >> 4, nq = idx & 15;
      rW[it] = nq * 4; cW[it] = k;
      wP[it] = Wb + (long)k * pp.ldw + (n0 + nq * 4);
    }
  }

  float4 pa[4], pw[4];

  auto loadGlb = [&]() {
#pragma unroll
    for (int it = 0; it < 4; ++it) {
      float4 v = make_float4(0.f, 0.f, 0.f, 0.f);
      if (aP[it]) { v = *(const float4*)aP[it]; aP[it] += BK; }
      pa[it] = v;
    }
    if (pp.wLayout == 0) {
#pragma unroll
      for (int it = 0; it < 4; ++it) {
        float4 v = make_float4(0.f, 0.f, 0.f, 0.f);
        if (wP[it]) { v = *(const float4*)wP[it]; wP[it] += BK; }
        pw[it] = v;
      }
    } else {
#pragma unroll
      for (int it = 0; it < 4; ++it) {
        float4 v = make_float4(0.f, 0.f, 0.f, 0.f);
        const float* p = wP[it];
        if (!edgeN) {
          v = *(const float4*)p;
        } else {
          int ng = n0 + rW[it];
          if (ng + 0 < pp.N) v.x = p[0];
          if (ng + 1 < pp.N) v.y = p[1];
          if (ng + 2 < pp.N) v.z = p[2];
          if (ng + 3 < pp.N) v.w = p[3];
        }
        wP[it] += wAdv;
        pw[it] = v;
      }
    }
  };

  auto storeLds = [&](int buf) {
#pragma unroll
    for (int it = 0; it < 4; ++it) {
      v4bf t;
      t[0] = (bf16_t)pa[it].x; t[1] = (bf16_t)pa[it].y;
      t[2] = (bf16_t)pa[it].z; t[3] = (bf16_t)pa[it].w;
      *(v4bf*)&As[buf][rA[it]][cA[it]] = t;
    }
    if (pp.wLayout == 0) {
#pragma unroll
      for (int it = 0; it < 4; ++it) {
        v4bf t;
        t[0] = (bf16_t)pw[it].x; t[1] = (bf16_t)pw[it].y;
        t[2] = (bf16_t)pw[it].z; t[3] = (bf16_t)pw[it].w;
        *(v4bf*)&Bs[buf][rW[it]][cW[it]] = t;
      }
    } else {
#pragma unroll
      for (int it = 0; it < 4; ++it) {
        Bs[buf][rW[it] + 0][cW[it]] = (bf16_t)pw[it].x;
        Bs[buf][rW[it] + 1][cW[it]] = (bf16_t)pw[it].y;
        Bs[buf][rW[it] + 2][cW[it]] = (bf16_t)pw[it].z;
        Bs[buf][rW[it] + 3][cW[it]] = (bf16_t)pw[it].w;
      }
    }
  };

  v8f acc[2][2];
#pragma unroll
  for (int a = 0; a < 2; ++a)
#pragma unroll
    for (int b = 0; b < 2; ++b)
#pragma unroll
      for (int i = 0; i < 8; ++i) acc[a][b][i] = 0.0f;

  auto compute = [&](int buf) {
    v16bf afr[2], bfr[2];
#pragma unroll
    for (int mi = 0; mi < 2; ++mi) {
      int r = wRow * 32 + mi * 16 + l16;
      v8bf c1 = *(const v8bf*)&As[buf][r][half * 8];
      v8bf c2 = *(const v8bf*)&As[buf][r][16 + half * 8];
#pragma unroll
      for (int j = 0; j < 8; ++j) { afr[mi][j] = c1[j]; afr[mi][8 + j] = c2[j]; }
    }
#pragma unroll
    for (int ni = 0; ni < 2; ++ni) {
      int c = wCol * 32 + ni * 16 + l16;
      v8bf c1 = *(const v8bf*)&Bs[buf][c][half * 8];
      v8bf c2 = *(const v8bf*)&Bs[buf][c][16 + half * 8];
#pragma unroll
      for (int j = 0; j < 8; ++j) { bfr[ni][j] = c1[j]; bfr[ni][8 + j] = c2[j]; }
    }
#pragma unroll
    for (int mi = 0; mi < 2; ++mi)
#pragma unroll
      for (int ni = 0; ni < 2; ++ni)
        acc[mi][ni] = __builtin_amdgcn_wmma_f32_16x16x32_bf16(
            false, afr[mi], false, bfr[ni], (short)0, acc[mi][ni], false, false);
  };

  // ---- double-buffered pipeline: one barrier per K-step, loads overlap WMMA
  const int nSteps = pp.K / BK;
  if (nSteps > 0) {
    loadGlb();
    storeLds(0);
    __syncthreads();
    for (int s = 0; s < nSteps; ++s) {
      const int cur = s & 1;
      const bool more = (s + 1) < nSteps;
      if (more) loadGlb();          // issue next tile's global loads early
      compute(cur);                 // WMMA on current buffer
      if (more) {
        storeLds(cur ^ 1);          // fill the other buffer
        __syncthreads();
      }
    }
  }

  // ---- epilogue ----
  const float* Rb = pp.res ? (pp.res + zq * pp.sC1 + (long)zr * pp.sC2) : nullptr;
#pragma unroll
  for (int mi = 0; mi < 2; ++mi) {
#pragma unroll
    for (int ni = 0; ni < 2; ++ni) {
#pragma unroll
      for (int i = 0; i < 8; ++i) {
        int m = m0 + wRow * 32 + mi * 16 + half * 8 + i;
        int n = n0 + wCol * 32 + ni * 16 + l16;
        if (m < pp.M && n < pp.N) {
          float v = acc[mi][ni][i] * pp.alpha;
          if (pp.bias) v += pp.bias[n];
          if (pp.act == 1) v = v > 0.f ? v : 0.01f * v;
          long ci = (long)m * pp.ldc + n;
          if (Rb) v += Rb[ci];
          if (pp.accum) v += Cb[ci];
          Cb[ci] = v;
        }
      }
    }
  }
}

// ---------------------------------------------------------------------------
// Elementwise / reduction kernels
// ---------------------------------------------------------------------------
__global__ __launch_bounds__(256) void embed_kernel(const int* __restrict__ x,
                                                    const float* __restrict__ emb,
                                                    float* __restrict__ out,
                                                    long total, int H) {
  long i = (long)blockIdx.x * blockDim.x + threadIdx.x;
  if (i < total) {
    long row = i / H;
    int  c   = (int)(i - row * H);
    out[i] = emb[(long)x[row] * H + c];
  }
}

__global__ __launch_bounds__(256) void mask_zero_kernel(const int* __restrict__ x,
                                                        float* __restrict__ h,
                                                        long total, int H) {
  long i = (long)blockIdx.x * blockDim.x + threadIdx.x;
  if (i < total) {
    long row = i / H;
    if (x[row] == 0) h[i] = 0.0f;
  }
}

// LayerNorm (optional affine, optional tanh); one block per row
__global__ __launch_bounds__(256) void ln_kernel(const float* __restrict__ x,
                                                 float* __restrict__ y,
                                                 const float* s, const float* b,
                                                 int n, int doTanh) {
  long row = blockIdx.x;
  const float* xr = x + row * n;
  float* yr = y + row * n;
  float sum = 0.f, sq = 0.f;
  for (int j = threadIdx.x; j < n; j += 256) { float v = xr[j]; sum += v; sq += v * v; }
  __shared__ float r1[256], r2[256];
  r1[threadIdx.x] = sum; r2[threadIdx.x] = sq;
  __syncthreads();
  for (int off = 128; off > 0; off >>= 1) {
    if (threadIdx.x < off) {
      r1[threadIdx.x] += r1[threadIdx.x + off];
      r2[threadIdx.x] += r2[threadIdx.x + off];
    }
    __syncthreads();
  }
  float mean = r1[0] / n;
  float var  = r2[0] / n - mean * mean;
  float inv  = rsqrtf(var + 1e-5f);
  for (int j = threadIdx.x; j < n; j += 256) {
    float v = (xr[j] - mean) * inv;
    if (s) v = v * s[j] + b[j];
    if (doTanh) v = tanhf(v);
    yr[j] = v;
  }
}

// Parser head softmax: p = softmax_j( mask_j ? logits*scale : -inf ), in place.
__global__ __launch_bounds__(256) void parser_softmax_kernel(float* __restrict__ p,
                                                             const int* __restrict__ x,
                                                             int L, float scale) {
  int row = blockIdx.x;              // b*L + i
  int b   = row / L;
  float* pr = p + (long)row * L;
  int j = threadIdx.x;
  __shared__ float red[256];
  float v = (x[b * L + j] != 0) ? pr[j] * scale : -INFINITY;
  red[j] = v; __syncthreads();
  for (int off = 128; off > 0; off >>= 1) {
    if (j < off) red[j] = fmaxf(red[j], red[j + off]);
    __syncthreads();
  }
  float mx = red[0]; __syncthreads();
  float e = (v == -INFINITY) ? 0.f : __expf(v - mx);
  red[j] = e; __syncthreads();
  for (int off = 128; off > 0; off >>= 1) {
    if (j < off) red[j] += red[j + off];
    __syncthreads();
  }
  pr[j] = e / red[0];
}

// Attention softmax with structural bias dep = w0*head + w1*head^T and key mask.
__global__ __launch_bounds__(256) void att_softmax_kernel(float* __restrict__ att,
                                                          const float* __restrict__ p,
                                                          const float* __restrict__ relw,
                                                          const int* __restrict__ x,
                                                          int L, int NH) {
  int row = blockIdx.x;              // ((b*NH+hd)*L + i)
  int i   = row % L;
  int bh  = row / L;
  int hd  = bh % NH;
  int b   = bh / NH;
  float r0 = relw[hd * 2 + 0], r1 = relw[hd * 2 + 1];
  float mr = fmaxf(r0, r1);
  float e0 = __expf(r0 - mr), e1 = __expf(r1 - mr);
  float w0 = e0 / (e0 + e1), w1 = e1 / (e0 + e1);

  float* ar = att + (long)row * L;
  int j = threadIdx.x;
  __shared__ float red[256];
  float ph = (j == i) ? 0.f : p[((long)b * L + i) * L + j];
  float pc = (j == i) ? 0.f : p[((long)b * L + j) * L + i];
  float v = (x[b * L + j] != 0) ? (ar[j] + w0 * ph + w1 * pc) : -INFINITY;
  red[j] = v; __syncthreads();
  for (int off = 128; off > 0; off >>= 1) {
    if (j < off) red[j] = fmaxf(red[j], red[j + off]);
    __syncthreads();
  }
  float mx = red[0]; __syncthreads();
  float e = (v == -INFINITY) ? 0.f : __expf(v - mx);
  red[j] = e; __syncthreads();
  for (int off = 128; off > 0; off >>= 1) {
    if (j < off) red[j] += red[j + off];
    __syncthreads();
  }
  ar[j] = e / red[0];
}

// ---------------------------------------------------------------------------
// Host-side helpers + orchestration
// ---------------------------------------------------------------------------
static void launch_gemm(hipStream_t st, const float* A, const float* W,
                        const float* bias, const float* res, float* C,
                        int M, int N, int K, int lda, int ldw, int ldc,
                        int wLayout, float alpha, int act, int accum,
                        int rowShift, int segLen,
                        long sA1, int sA2, long sW1, int sW2, long sC1, int sC2,
                        int zdiv, int batch) {
  GemmP p;
  p.A = A; p.W = W; p.bias = bias; p.res = res; p.C = C;
  p.M = M; p.N = N; p.K = K;
  p.lda = lda; p.ldw = ldw; p.ldc = ldc;
  p.wLayout = wLayout; p.alpha = alpha; p.act = act; p.accum = accum;
  p.rowShift = rowShift; p.segLen = (segLen > 0 ? segLen : 1);
  p.sA1 = sA1; p.sA2 = sA2; p.sW1 = sW1; p.sW2 = sW2; p.sC1 = sC1; p.sC2 = sC2;
  p.zdiv = (zdiv > 0 ? zdiv : 1);
  dim3 g((N + BN - 1) / BN, (M + BM - 1) / BM, batch);
  hipLaunchKernelGGL(gemm_wmma_bf16, g, dim3(GT), 0, st, p);
}

extern "C" void kernel_launch(void* const* d_in, const int* in_sizes, int n_in,
                              void* d_out, int out_size, void* d_ws, size_t ws_size,
                              hipStream_t stream) {
  (void)in_sizes; (void)n_in; (void)out_size; (void)ws_size;
  const int B = 16, L = 256, H = 512, NH = 8, NL = 6, NTOK = 10000, P = 4, KK = 9;
  const int DFF = 4 * H;
  const int M = B * L;
  const int DH = H / NH;
  const float scale   = 1.0f / sqrtf((float)H);
  const float dhscale = 1.0f / sqrtf((float)DH);

  const int*   x        = (const int*)  d_in[0];
  const float* emb      = (const float*)d_in[2];
  const float* conv_w   = (const float*)d_in[3];
  const float* conv_b   = (const float*)d_in[4];
  const float* parent_w = (const float*)d_in[5];
  const float* parent_b = (const float*)d_in[6];
  const float* child_w  = (const float*)d_in[7];
  const float* child_b  = (const float*)d_in[8];
  const float* rel_w    = (const float*)d_in[9];
  const float* qkv_w    = (const float*)d_in[10];
  const float* qkv_b    = (const float*)d_in[11];
  const float* out_w    = (const float*)d_in[12];
  const float* out_b    = (const float*)d_in[13];
  const float* ln1_s    = (const float*)d_in[14];
  const float* ln1_b    = (const float*)d_in[15];
  const float* ln2_s    = (const float*)d_in[16];
  const float* ln2_b    = (const float*)d_in[17];
  const float* ff1_w    = (const float*)d_in[18];
  const float* ff1_b    = (const float*)d_in[19];
  const float* ff2_w    = (const float*)d_in[20];
  const float* ff2_b    = (const float*)d_in[21];
  const float* norm_s   = (const float*)d_in[22];
  const float* norm_b   = (const float*)d_in[23];
  const float* out_bias = (const float*)d_in[24];

  // workspace layout (floats)
  float* ws    = (float*)d_ws;
  float* f_h   = ws;                                  // M*H
  float* f_hn  = f_h  + (long)M * H;                  // M*H
  float* f_qkv = f_hn + (long)M * H;                  // M*3H
  float* f_att = f_qkv + (long)M * 3 * H;             // B*NH*L*L
  float* f_ctx = f_att + (long)B * NH * L * L;        // M*H
  float* f_mid = f_ctx + (long)M * H;                 // M*DFF
  float* f_p   = f_mid + (long)M * DFF;               // B*L*L
  // parser-phase aliases (freed before transformer uses them)
  float* hp  = f_qkv;
  float* tmp = f_att;
  float* par = f_mid;
  float* chi = f_mid + (long)M * H;

  const long total = (long)M * H;
  const int  eb    = (int)((total + 255) / 256);

  // ---- differentiable parser: 4x (mask -> conv(9 shifted GEMMs) -> LN -> tanh)
  hipLaunchKernelGGL(embed_kernel, dim3(eb), dim3(256), 0, stream, x, emb, hp, total, H);
  for (int pi = 0; pi < P; ++pi) {
    hipLaunchKernelGGL(mask_zero_kernel, dim3(eb), dim3(256), 0, stream, x, hp, total, H);
    for (int t = 0; t < KK; ++t) {
      launch_gemm(stream, hp, conv_w + ((long)pi * KK + t) * H * H,
                  (t == 0) ? (conv_b + pi * H) : nullptr, nullptr, tmp,
                  M, H, H, H, H, H, /*wLayout*/1, 1.0f, 0, /*accum*/(t == 0) ? 0 : 1,
                  /*rowShift*/t - 4, /*segLen*/L, 0, 0, 0, 0, 0, 0, 1, 1);
    }
    hipLaunchKernelGGL(ln_kernel, dim3(M), dim3(256), 0, stream,
                       tmp, hp, (const float*)nullptr, (const float*)nullptr, H, 1);
  }
  // parent / child projections
  launch_gemm(stream, hp, parent_w, parent_b, nullptr, par,
              M, H, H, H, H, H, 0, 1.0f, 0, 0, 0, 0, 0, 0, 0, 0, 0, 0, 1, 1);
  launch_gemm(stream, hp, child_w, child_b, nullptr, chi,
              M, H, H, H, H, H, 0, 1.0f, 0, 0, 0, 0, 0, 0, 0, 0, 0, 0, 1, 1);
  // logits[b] = child[b] @ parent[b]^T * scale   (batched over B)
  launch_gemm(stream, chi, par, nullptr, nullptr, f_p,
              L, L, H, H, H, L, 0, scale, 0, 0, 0, 0,
              (long)L * H, 0, (long)L * H, 0, (long)L * L, 0, 1, B);
  hipLaunchKernelGGL(parser_softmax_kernel, dim3(B * L), dim3(256), 0, stream,
                     f_p, x, L, scale);

  // ---- transformer encoder ----
  hipLaunchKernelGGL(embed_kernel, dim3(eb), dim3(256), 0, stream, x, emb, f_h, total, H);
  for (int i = 0; i < NL; ++i) {
    hipLaunchKernelGGL(ln_kernel, dim3(M), dim3(256), 0, stream,
                       f_h, f_hn, ln1_s + (long)i * H, ln1_b + (long)i * H, H, 0);
    launch_gemm(stream, f_hn, qkv_w + (long)i * 3 * H * H, qkv_b + (long)i * 3 * H,
                nullptr, f_qkv, M, 3 * H, H, H, H, 3 * H, 0, 1.0f, 0, 0, 0, 0,
                0, 0, 0, 0, 0, 0, 1, 1);
    // scores: q @ k^T * DH^-0.5, batched over (b, head)
    launch_gemm(stream, f_qkv, f_qkv + H, nullptr, nullptr, f_att,
                L, L, DH, 3 * H, 3 * H, L, 0, dhscale, 0, 0, 0, 0,
                (long)L * 3 * H, DH, (long)L * 3 * H, DH,
                (long)NH * L * L, L * L, NH, B * NH);
    hipLaunchKernelGGL(att_softmax_kernel, dim3(B * NH * L), dim3(256), 0, stream,
                       f_att, f_p, rel_w + (long)i * NH * 2, x, L, NH);
    // ctx: att @ v
    launch_gemm(stream, f_att, f_qkv + 2 * H, nullptr, nullptr, f_ctx,
                L, DH, L, L, 3 * H, H, 1, 1.0f, 0, 0, 0, 0,
                (long)NH * L * L, L * L, (long)L * 3 * H, DH,
                (long)L * H, DH, NH, B * NH);
    // h = h + ctx @ out_w^T + out_b
    launch_gemm(stream, f_ctx, out_w + (long)i * H * H, out_b + (long)i * H,
                f_h, f_h, M, H, H, H, H, H, 0, 1.0f, 0, 0, 0, 0,
                0, 0, 0, 0, 0, 0, 1, 1);
    hipLaunchKernelGGL(ln_kernel, dim3(M), dim3(256), 0, stream,
                       f_h, f_hn, ln2_s + (long)i * H, ln2_b + (long)i * H, H, 0);
    // mid = leaky_relu(hn @ ff1^T + b1)
    launch_gemm(stream, f_hn, ff1_w + (long)i * DFF * H, ff1_b + (long)i * DFF,
                nullptr, f_mid, M, DFF, H, H, H, DFF, 0, 1.0f, 1, 0, 0, 0,
                0, 0, 0, 0, 0, 0, 1, 1);
    // h = h + mid @ ff2^T + b2
    launch_gemm(stream, f_mid, ff2_w + (long)i * H * DFF, ff2_b + (long)i * H,
                f_h, f_h, M, H, DFF, DFF, DFF, H, 0, 1.0f, 0, 0, 0, 0,
                0, 0, 0, 0, 0, 0, 1, 1);
  }

  // ---- final LN + tied-embedding output projection ----
  hipLaunchKernelGGL(ln_kernel, dim3(M), dim3(256), 0, stream,
                     f_h, f_hn, norm_s, norm_b, H, 0);
  launch_gemm(stream, f_hn, emb, out_bias, nullptr, (float*)d_out,
              M, NTOK, H, H, H, NTOK, 0, 1.0f, 0, 0, 0, 0,
              0, 0, 0, 0, 0, 0, 1, 1);
}
